// MultiHeadSelfAttention_35845797053009
// MI455X (gfx1250) — matrix operand order repaired and checked by reading, compiled, not run on hardware
//
#include <hip/hip_runtime.h>

// MI455X (gfx1250) fused multi-head self-attention.
// Pipeline: cvt(x,W* -> f16) -> [xh@W{q,k,v} + b -> f16] -> flash attention
//           (online softmax, f32 stats) -> f16 -> [attn@Wo + bo -> f32 out]
// All matmuls use v_wmma_f32_16x16x32_f16 (fp32 accumulate).

typedef __attribute__((ext_vector_type(16))) _Float16 v16h;
typedef __attribute__((ext_vector_type(4)))  _Float16 v4h;
typedef __attribute__((ext_vector_type(8)))  float    v8f;

#define WMMA_F16(a, b, c) \
  __builtin_amdgcn_wmma_f32_16x16x32_f16(false, (a), false, (b), (short)0, (c), false, false)

// A-matrix 16x32 f16 fragment (ISA layout): lane L holds row M=L&15, half h=L>>4.
// halves 0..7  -> K = 8h .. 8h+7        (16B at p)
// halves 8..15 -> K = 16+8h .. 16+8h+7  (16B at p+16 halves)
__device__ __forceinline__ v16h frag_ld_a(const _Float16* p) {
  union { v16h h; uint4 q[2]; } u;
  u.q[0] = *reinterpret_cast<const uint4*>(p);
  u.q[1] = *reinterpret_cast<const uint4*>(p + 16);
  return u.h;
}
// B-matrix 32x16 f16 fragment: lane L holds col N=L&15; halves 0..7 -> K = 8h..8h+7,
// halves 8..15 -> K = 16+8h..16+8h+7 (two 16B loads from a [col][k] LDS layout).
__device__ __forceinline__ v16h frag_ld_b(const _Float16* p) {
  union { v16h h; uint4 q[2]; } u;
  u.q[0] = *reinterpret_cast<const uint4*>(p);
  u.q[1] = *reinterpret_cast<const uint4*>(p + 16);
  return u.h;
}

// ---------------------------------------------------------------------------
// f32 -> f16 conversion (vectorized, one shot for x and the four weights).
// ---------------------------------------------------------------------------
__global__ __launch_bounds__(256) void cvt_f32_to_f16_kernel(
    const float* __restrict__ src, _Float16* __restrict__ dst, int n4) {
  int i = blockIdx.x * blockDim.x + threadIdx.x;
  if (i < n4) {
    float4 f = reinterpret_cast<const float4*>(src)[i];
    v4h h = {(_Float16)f.x, (_Float16)f.y, (_Float16)f.z, (_Float16)f.w};
    reinterpret_cast<v4h*>(dst)[i] = h;
  }
}

// ---------------------------------------------------------------------------
// GEMM: C[M,N] = A[M,K] @ B[K,N] + bias. A,B: f16. C: f16 or f32 (template).
// Block tile 128x128x32, 256 threads = 8 wave32 arranged 4(M) x 2(N);
// each wave computes 32x64 = 2x4 WMMA accumulators.
// ---------------------------------------------------------------------------
template <bool C_F16>
__global__ __launch_bounds__(256) void gemm_wmma_kernel(
    const _Float16* __restrict__ Ag, const _Float16* __restrict__ Bg,
    const float* __restrict__ bias, void* __restrict__ Cg,
    int M, int N, int K) {
  constexpr int BM = 128, BN = 128, BK = 32, LDT = 56;  // 56 halves = 112B stride
  __shared__ __align__(16) _Float16 As[BM * LDT];  // [m][k]
  __shared__ __align__(16) _Float16 Bs[BN * LDT];  // [n][k] (transposed)

  const int t   = threadIdx.x;
  const int wid = t >> 5, lane = t & 31;
  const int l15 = lane & 15, hh = lane >> 4;
  const int wm  = wid & 3, wn = wid >> 2;
  const int m0  = blockIdx.x * BM, n0 = blockIdx.y * BN;

  v8f vzero = {};
  v8f acc[2][4];
#pragma unroll
  for (int i = 0; i < 2; ++i)
#pragma unroll
    for (int j = 0; j < 4; ++j) acc[i][j] = vzero;

  for (int kk = 0; kk < K; kk += BK) {
    __syncthreads();
    // ---- stage A tile: pure 16B copies ----
#pragma unroll
    for (int p = 0; p < 2; ++p) {
      int lin = p * 2048 + t * 8;
      int row = lin >> 5, k = lin & 31;
      uint4 v = *reinterpret_cast<const uint4*>(Ag + (size_t)(m0 + row) * K + kk + k);
      *reinterpret_cast<uint4*>(&As[row * LDT + k]) = v;
    }
    // ---- stage B tile transposed: 16B coalesced loads, b16 scatter ----
#pragma unroll
    for (int p = 0; p < 2; ++p) {
      int idx = p * 256 + t;        // 0..511
      int k = idx >> 4;             // 0..31
      int n8 = (idx & 15) * 8;
      union { uint4 q; _Float16 h[8]; } u;
      u.q = *reinterpret_cast<const uint4*>(Bg + (size_t)(kk + k) * N + n0 + n8);
#pragma unroll
      for (int i = 0; i < 8; ++i) Bs[(n8 + i) * LDT + k] = u.h[i];
    }
    // prefetch next K-step tiles into cache (global_prefetch_b8; no counters)
    if (kk + BK < K) {
      __builtin_prefetch(Ag + (size_t)(m0 + (t >> 3)) * K + kk + BK, 0, 1);
      __builtin_prefetch(Bg + (size_t)(kk + BK + (t >> 4)) * N + n0 + (t & 15) * 8, 0, 1);
    }
    __syncthreads();

    v16h af[2], bf[4];
#pragma unroll
    for (int mt = 0; mt < 2; ++mt)
      af[mt] = frag_ld_a(&As[(wm * 32 + mt * 16 + l15) * LDT + 8 * hh]);
#pragma unroll
    for (int nt = 0; nt < 4; ++nt)
      bf[nt] = frag_ld_b(&Bs[(wn * 64 + nt * 16 + l15) * LDT + 8 * hh]);
#pragma unroll
    for (int mt = 0; mt < 2; ++mt)
#pragma unroll
      for (int nt = 0; nt < 4; ++nt)
        acc[mt][nt] = WMMA_F16(af[mt], bf[nt], acc[mt][nt]);
  }

  // epilogue: C layout — VGPR r, lane: N = l15, M = r + 8*hh
#pragma unroll
  for (int mt = 0; mt < 2; ++mt)
#pragma unroll
    for (int nt = 0; nt < 4; ++nt) {
      const int n = n0 + wn * 64 + nt * 16 + l15;
      const float bv = bias[n];
#pragma unroll
      for (int r = 0; r < 8; ++r) {
        const int m = m0 + wm * 32 + mt * 16 + r + 8 * hh;
        float v = acc[mt][nt][r] + bv;
        if (C_F16) ((_Float16*)Cg)[(size_t)m * N + n] = (_Float16)v;
        else       ((float*)Cg)[(size_t)m * N + n] = v;
      }
    }
}

// ---------------------------------------------------------------------------
// Flash attention: one WG (256 thr, 8 waves) per (n, head, 128 q-rows).
// Streams 64-key blocks: S = (Q Kt)/8, mask, online softmax, O += P V.
// Mask staged TRANSPOSED [key][q] so each lane reads its 8 q-rows of one key
// column as a single 8-byte LDS load (4 x ds_load_b64 instead of 32 x b8).
// ---------------------------------------------------------------------------
__global__ __launch_bounds__(256) void attn_flash_kernel(
    const _Float16* __restrict__ Qh, const _Float16* __restrict__ Kh,
    const _Float16* __restrict__ Vh, const int* __restrict__ mask,
    _Float16* __restrict__ AO) {
  constexpr int S = 2048, D = 1024, HD = 64, QB = 128, KB = 64;
  constexpr int QLD = 72, KLD = 72, VLD = 72, PLD = 72;  // 144B strides
  constexpr int MLD = 136;                               // mask row stride (8B-aligned)
  __shared__ __align__(16) _Float16 Qs[QB * QLD];
  __shared__ __align__(16) _Float16 Ks[KB * KLD];   // [key][d]
  __shared__ __align__(16) _Float16 Vs[HD * VLD];   // [d][key] (transposed)
  __shared__ __align__(16) _Float16 Ps[QB * PLD];   // [q][key]
  __shared__ __align__(8)  unsigned char Ms[KB * MLD];  // [key][q] transposed

  const int t   = threadIdx.x;
  const int wid = t >> 5, lane = t & 31;
  const int l15 = lane & 15, hh = lane >> 4;
  const int n = blockIdx.z, head = blockIdx.y;
  const int qb0 = blockIdx.x * QB;

  const _Float16* Qg = Qh + ((size_t)n * S + qb0) * D + head * HD;
  const _Float16* Kg = Kh + (size_t)n * S * D + head * HD;
  const _Float16* Vg = Vh + (size_t)n * S * D + head * HD;
  const int* Mg = mask + (size_t)n * S * S;

  // ---- stage Q (128 x 64 halves), load per-wave A fragments, keep in regs ----
  {
    int row = t >> 1, c = (t & 1) * 32;
    const uint4* src = reinterpret_cast<const uint4*>(Qg + (size_t)row * D + c);
    uint4* dst = reinterpret_cast<uint4*>(&Qs[row * QLD + c]);
    dst[0] = src[0]; dst[1] = src[1]; dst[2] = src[2]; dst[3] = src[3];
  }
  __syncthreads();
  v16h aq[2];
#pragma unroll
  for (int ds = 0; ds < 2; ++ds)
    aq[ds] = frag_ld_a(&Qs[(wid * 16 + l15) * QLD + ds * 32 + 8 * hh]);

  v8f vzero = {};
  v8f o[4];
  float m_r[8], l_r[8];
#pragma unroll
  for (int i = 0; i < 4; ++i) o[i] = vzero;
#pragma unroll
  for (int r = 0; r < 8; ++r) { m_r[r] = -1e30f; l_r[r] = 0.f; }

  for (int kb0 = 0; kb0 < S; kb0 += KB) {
    __syncthreads();  // previous iteration done consuming Ks/Vs/Ps/Ms
    {  // stage K row-major and V transposed (16 halves per thread each)
      int row = t >> 2, c = (t & 3) * 16;
      const uint4* ks = reinterpret_cast<const uint4*>(Kg + (size_t)(kb0 + row) * D + c);
      uint4* kd = reinterpret_cast<uint4*>(&Ks[row * KLD + c]);
      kd[0] = ks[0]; kd[1] = ks[1];
      union { uint4 q[2]; _Float16 h[16]; } tv;
      const uint4* vs = reinterpret_cast<const uint4*>(Vg + (size_t)(kb0 + row) * D + c);
      tv.q[0] = vs[0]; tv.q[1] = vs[1];
#pragma unroll
      for (int i = 0; i < 16; ++i) Vs[(c + i) * VLD + row] = tv.h[i];
      // prefetch next key-block K/V rows (global_prefetch_b8)
      if (kb0 + KB < S) {
        __builtin_prefetch(Kg + (size_t)(kb0 + KB + row) * D + c, 0, 1);
        __builtin_prefetch(Vg + (size_t)(kb0 + KB + row) * D + c, 0, 1);
      }
    }
#pragma unroll
    for (int p = 0; p < 32; ++p) {  // stage mask transposed (coalesced reads over k)
      int idx = p * 256 + t;
      int ql = idx >> 6, kl = idx & 63;
      Ms[kl * MLD + ql] = (unsigned char)(Mg[(size_t)(qb0 + ql) * S + kb0 + kl] != 0);
    }
    if (kb0 + KB < S)
      __builtin_prefetch(Mg + (size_t)(qb0 + (t >> 1)) * S + kb0 + KB, 0, 1);
    __syncthreads();

    // ---- scores: 16(q) x 64(key) per wave = 4 n-tiles x 2 d-steps of WMMA ----
    v8f sc[4];
#pragma unroll
    for (int nt = 0; nt < 4; ++nt) {
      v8f s = vzero;
#pragma unroll
      for (int ds = 0; ds < 2; ++ds) {
        v16h bk = frag_ld_b(&Ks[(nt * 16 + l15) * KLD + ds * 32 + 8 * hh]);
        s = WMMA_F16(aq[ds], bk, s);
      }
      sc[nt] = s;
    }

    // mask bits for this lane's key columns: 8 q-rows per 8-byte load
    unsigned long long mb[4];
#pragma unroll
    for (int nt = 0; nt < 4; ++nt)
      mb[nt] = *reinterpret_cast<const unsigned long long*>(
          &Ms[(nt * 16 + l15) * MLD + wid * 16 + 8 * hh]);

    // ---- mask + scale (1/sqrt(64)) + online softmax ----
#pragma unroll
    for (int r = 0; r < 8; ++r) {
      const int qrow = wid * 16 + r + 8 * hh;
      float mx = -1e30f;
#pragma unroll
      for (int nt = 0; nt < 4; ++nt) {
        float v = sc[nt][r] * 0.125f;
        if (((mb[nt] >> (8 * r)) & 1ull) == 0) v = -1e30f;
        sc[nt][r] = v;
        mx = fmaxf(mx, v);
      }
#pragma unroll
      for (int off = 1; off < 16; off <<= 1)
        mx = fmaxf(mx, __shfl_xor(mx, off, 16));  // row max over 16 lanes
      const float mnew = fmaxf(m_r[r], mx);
      const float alpha = __expf(m_r[r] - mnew);
      m_r[r] = mnew;
      float rsum = 0.f;
#pragma unroll
      for (int nt = 0; nt < 4; ++nt) {
        float sv = sc[nt][r];
        float p = (sv < -1e29f) ? 0.f : __expf(sv - mnew);  // masked -> exactly 0
        Ps[qrow * PLD + nt * 16 + l15] = (_Float16)p;
        rsum += p;
      }
#pragma unroll
      for (int off = 1; off < 16; off <<= 1)
        rsum += __shfl_xor(rsum, off, 16);
      l_r[r] = l_r[r] * alpha + rsum;
#pragma unroll
      for (int dt = 0; dt < 4; ++dt) o[dt][r] *= alpha;  // rescale O
    }
    __syncthreads();  // Ps visible

    // ---- O += P @ V : 2 k-steps x 4 d-tiles of WMMA ----
#pragma unroll
    for (int ks = 0; ks < 2; ++ks) {
      v16h ap = frag_ld_a(&Ps[(wid * 16 + l15) * PLD + ks * 32 + 8 * hh]);
#pragma unroll
      for (int dt = 0; dt < 4; ++dt) {
        v16h bv = frag_ld_b(&Vs[(dt * 16 + l15) * VLD + ks * 32 + 8 * hh]);
        o[dt] = WMMA_F16(ap, bv, o[dt]);
      }
    }
  }

  // ---- normalize and write attn output (f16, [N,S,D] merged-head layout) ----
#pragma unroll
  for (int dt = 0; dt < 4; ++dt)
#pragma unroll
    for (int r = 0; r < 8; ++r) {
      float denom = l_r[r];
      denom = (denom > 1e-30f) ? denom : 1.f;  // fully-masked-row guard
      float val = o[dt][r] / denom;
      AO[((size_t)n * S + qb0 + wid * 16 + r + 8 * hh) * D + head * HD + dt * 16 + l15] =
          (_Float16)val;
    }
}

// ---------------------------------------------------------------------------
extern "C" void kernel_launch(void* const* d_in, const int* in_sizes, int n_in,
                              void* d_out, int out_size, void* d_ws, size_t ws_size,
                              hipStream_t stream) {
  constexpr int N = 2, S = 2048, D = 1024;
  constexpr int M = N * S;  // 4096 token rows

  const float* x    = (const float*)d_in[0];
  const int*   mask = (const int*)d_in[1];
  const float* Wq = (const float*)d_in[2];
  const float* bq = (const float*)d_in[3];
  const float* Wk = (const float*)d_in[4];
  const float* bk = (const float*)d_in[5];
  const float* Wv = (const float*)d_in[6];
  const float* bv = (const float*)d_in[7];
  const float* Wo = (const float*)d_in[8];
  const float* bo = (const float*)d_in[9];
  float* out = (float*)d_out;

  // Workspace (f16): xh, Qh, Kh, Vh, AO = 4M halves each; Wqh..Woh = 1M each.
  _Float16* ws = (_Float16*)d_ws;
  const size_t tok = (size_t)M * D;  // 4M
  const size_t wsz = (size_t)D * D;  // 1M
  _Float16* xh  = ws;
  _Float16* Qh  = ws + tok;
  _Float16* Kh  = ws + 2 * tok;
  _Float16* Vh  = ws + 3 * tok;
  _Float16* AO  = ws + 4 * tok;
  _Float16* Wqh = ws + 5 * tok;
  _Float16* Wkh = Wqh + wsz;
  _Float16* Wvh = Wkh + wsz;
  _Float16* Woh = Wvh + wsz;

  dim3 blk(256);
  // one-shot f32 -> f16 conversions
  cvt_f32_to_f16_kernel<<<dim3((int)(tok / 4 / 256)), blk, 0, stream>>>(x, xh, (int)(tok / 4));
  cvt_f32_to_f16_kernel<<<dim3((int)(wsz / 4 / 256)), blk, 0, stream>>>(Wq, Wqh, (int)(wsz / 4));
  cvt_f32_to_f16_kernel<<<dim3((int)(wsz / 4 / 256)), blk, 0, stream>>>(Wk, Wkh, (int)(wsz / 4));
  cvt_f32_to_f16_kernel<<<dim3((int)(wsz / 4 / 256)), blk, 0, stream>>>(Wv, Wvh, (int)(wsz / 4));
  cvt_f32_to_f16_kernel<<<dim3((int)(wsz / 4 / 256)), blk, 0, stream>>>(Wo, Woh, (int)(wsz / 4));

  dim3 gg(M / 128, D / 128);  // 32 x 8 tiles
  gemm_wmma_kernel<true><<<gg, blk, 0, stream>>>(xh, Wqh, bq, Qh, M, D, D);
  gemm_wmma_kernel<true><<<gg, blk, 0, stream>>>(xh, Wkh, bk, Kh, M, D, D);
  gemm_wmma_kernel<true><<<gg, blk, 0, stream>>>(xh, Wvh, bv, Vh, M, D, D);

  attn_flash_kernel<<<dim3(S / 128, 16, N), blk, 0, stream>>>(Qh, Kh, Vh, mask, AO);

  gemm_wmma_kernel<false><<<gg, blk, 0, stream>>>(AO, Woh, bo, out, M, D, D);
}
